// CausalSelfAttention_738734375717
// MI455X (gfx1250) — matrix operand order repaired and checked by reading, not compiled, not run on hardware
//
#include <hip/hip_runtime.h>

#define TSEQ 2048
#define CDIM 1024
#define HEADS 16
#define HDIM 64
#define BATCH 2

typedef unsigned int   u32;
typedef unsigned short u16;
typedef __attribute__((ext_vector_type(16))) __bf16 v16bf;
typedef __attribute__((ext_vector_type(8)))  float  v8f;
typedef __attribute__((ext_vector_type(4)))  u32    u32x4;

union BfOp {
    u32x4 q[2];
    v16bf v;
};

__device__ __forceinline__ u16 cvt_bf16(float f) {
    u32 u = __float_as_uint(f);
    u += 0x7FFFu + ((u >> 16) & 1u);   // round-to-nearest-even
    return (u16)(u >> 16);
}

// 16 contiguous bf16 (B-operand per-lane chunk): two b128 loads
__device__ __forceinline__ v16bf ld_contig16(const u16* p) {
    BfOp b;
    b.q[0] = *(const u32x4*)(p);
    b.q[1] = *(const u32x4*)(p + 8);
    return b.v;
}

// A-operand per-lane chunk: K segment {0..15} relative to p
__device__ __forceinline__ v16bf ld_aseg(const u16* p) {
    BfOp b;
    b.q[0] = *(const u32x4*)(p);
    b.q[1] = *(const u32x4*)(p + 8);
    return b.v;
}

__device__ __forceinline__ v8f wmma_bf16(v16bf a, v16bf b, v8f c) {
    // (neg_a, A, neg_b, B, c_mod, C, reuse_a, reuse_b)
    return __builtin_amdgcn_wmma_f32_16x16x32_bf16(false, a, false, b, (short)0, c,
                                                   false, false);
}

#define V8ZERO {0.f,0.f,0.f,0.f,0.f,0.f,0.f,0.f}

// ---------------------------------------------------------------------------
// fp32 -> bf16 elementwise convert
// ---------------------------------------------------------------------------
__global__ void cvt_f32_bf16(const float* __restrict__ in, u16* __restrict__ out, int n) {
    int i = blockIdx.x * blockDim.x + threadIdx.x;
    if (i < n) out[i] = cvt_bf16(in[i]);
}

// ---------------------------------------------------------------------------
// VT[b,h,d,t] = V[b,t,h,d]  (V = qkv section at offset 2*CDIM)
// ---------------------------------------------------------------------------
__global__ void build_vt(const u16* __restrict__ qkv, u16* __restrict__ vt, int n) {
    int i = blockIdx.x * blockDim.x + threadIdx.x;
    if (i >= n) return;
    int t = i & (TSEQ - 1);
    int r = i >> 11;
    int d = r & (HDIM - 1);  r >>= 6;
    int h = r & (HEADS - 1);
    int b = r >> 4;
    vt[i] = qkv[(size_t)(b * TSEQ + t) * (3 * CDIM) + 2 * CDIM + h * HDIM + d];
}

// ---------------------------------------------------------------------------
// GEMM: C[M,N] = A[M,K] * B[N,K]^T   (A,B bf16 row-major K-contiguous)
// Wave computes 16x128; block = 4 waves -> 16x512 per block.
// ---------------------------------------------------------------------------
__global__ __launch_bounds__(128) void gemm_bf16_nt(
    const u16* __restrict__ A, const u16* __restrict__ Bw,
    float* __restrict__ Cf, u16* __restrict__ Cb,
    int M, int N, int K, int bf16out)
{
    const int lane = threadIdx.x & 31;
    const int wave = threadIdx.x >> 5;
    const int row  = lane & 15;
    const int hsel = lane >> 4;          // 0: lanes 0-15, 1: lanes 16-31
    const int m0 = blockIdx.x * 16;
    const int n0 = blockIdx.y * 512 + wave * 128;

    const u16* aRow = A  + (size_t)(m0 + row) * K + hsel * 16;
    const u16* bRow = Bw + (size_t)(n0 + row) * K + hsel * 16;

    v8f acc[8];
#pragma unroll
    for (int i = 0; i < 8; ++i) acc[i] = (v8f)V8ZERO;

    for (int k0 = 0; k0 < K; k0 += 32) {
        v16bf a = ld_aseg(aRow + k0);
#pragma unroll
        for (int nt = 0; nt < 8; ++nt) {
            v16bf b = ld_contig16(bRow + (size_t)nt * 16 * K + k0);
            acc[nt] = wmma_bf16(a, b, acc[nt]);
        }
    }

#pragma unroll
    for (int nt = 0; nt < 8; ++nt) {
#pragma unroll
        for (int r = 0; r < 8; ++r) {
            int gm = m0 + r + hsel * 8;          // C/D layout: M = r + 8*upperhalf
            int gn = n0 + nt * 16 + row;         // N = lane&15
            if (bf16out) Cb[(size_t)gm * N + gn] = cvt_bf16(acc[nt][r]);
            else         Cf[(size_t)gm * N + gn] = acc[nt][r];
        }
    }
}

// ---------------------------------------------------------------------------
// Fused causal flash attention. 1 wave / block; block = (b,h,qtile of 16).
// qkv: bf16 [B,T,3C];  vt: bf16 [B,H,D,T];  ob: bf16 [B,T,C]
// ---------------------------------------------------------------------------
__global__ __launch_bounds__(32) void attn_fwd(
    const u16* __restrict__ qkv, const u16* __restrict__ vt, u16* __restrict__ ob)
{
    const int lane = threadIdx.x;
    const int row  = lane & 15;
    const int hsel = lane >> 4;
    const int qt = blockIdx.x, h = blockIdx.y, b = blockIdx.z;
    const int q0 = qt * 16;

    // Q tile (16x64) as two A-operands (d 0..31 / 32..63), loaded once
    const u16* qrow = qkv + (size_t)(b * TSEQ + q0 + row) * (3 * CDIM) + h * HDIM + hsel * 16;
    v16bf qa0 = ld_aseg(qrow);
    v16bf qa1 = ld_aseg(qrow + 32);

    v8f o0 = (v8f)V8ZERO, o1 = (v8f)V8ZERO, o2 = (v8f)V8ZERO, o3 = (v8f)V8ZERO;
    float mrun[8], lrun[8];
#pragma unroll
    for (int r = 0; r < 8; ++r) { mrun[r] = -3.0e38f; lrun[r] = 0.0f; }

    __shared__ u16 P[16 * 40];   // 16 queries x 32 keys, padded stride 40 (16B aligned)

    const u16* kbase = qkv + (size_t)b * TSEQ * (3 * CDIM) + CDIM + h * HDIM + hsel * 16;
    const u16* vbase = vt + ((size_t)(b * HEADS + h) * HDIM + row) * TSEQ + hsel * 16;

    for (int kv0 = 0; kv0 < q0 + 16; kv0 += 32) {
        // ---- S = Q K^T : two 16x16 score tiles (keys kv0.. / kv0+16..)
        v8f s0 = (v8f)V8ZERO, s1 = (v8f)V8ZERO;
        const u16* k0p = kbase + (size_t)(kv0 + row) * (3 * CDIM);
        const u16* k1p = k0p + (size_t)16 * (3 * CDIM);
        s0 = wmma_bf16(qa0, ld_contig16(k0p),      s0);
        s0 = wmma_bf16(qa1, ld_contig16(k0p + 32), s0);
        s1 = wmma_bf16(qa0, ld_contig16(k1p),      s1);
        s1 = wmma_bf16(qa1, ld_contig16(k1p + 32), s1);

        // ---- causal mask + online softmax (row m = r + 8*hsel, spread over 16 lanes)
#pragma unroll
        for (int r = 0; r < 8; ++r) {
            int qm = q0 + r + hsel * 8;
            float v0 = (kv0 + row      <= qm) ? s0[r] * 0.125f : -3.0e38f;
            float v1 = (kv0 + 16 + row <= qm) ? s1[r] * 0.125f : -3.0e38f;
            float mt = fmaxf(v0, v1);
#pragma unroll
            for (int off = 8; off > 0; off >>= 1)
                mt = fmaxf(mt, __shfl_xor(mt, off, 32));   // reduce within 16-lane half
            float mnew  = fmaxf(mrun[r], mt);
            float alpha = __expf(mrun[r] - mnew);
            float p0 = __expf(v0 - mnew);
            float p1 = __expf(v1 - mnew);
            float ts = p0 + p1;
#pragma unroll
            for (int off = 8; off > 0; off >>= 1)
                ts += __shfl_xor(ts, off, 32);
            lrun[r] = lrun[r] * alpha + ts;
            mrun[r] = mnew;
            o0[r] *= alpha; o1[r] *= alpha; o2[r] *= alpha; o3[r] *= alpha;
            int prow = r + hsel * 8;
            P[prow * 40 + row]      = cvt_bf16(p0);
            P[prow * 40 + 16 + row] = cvt_bf16(p1);
        }
        asm volatile("s_wait_dscnt 0" ::: "memory");       // LDS store -> load

        // ---- reload P (16x32) in A-operand layout, then O += P V
        v16bf pa = ld_aseg(&P[row * 40 + hsel * 16]);
        const u16* vp = vbase + kv0;
        o0 = wmma_bf16(pa, ld_contig16(vp),             o0);
        o1 = wmma_bf16(pa, ld_contig16(vp + 16 * TSEQ), o1);
        o2 = wmma_bf16(pa, ld_contig16(vp + 32 * TSEQ), o2);
        o3 = wmma_bf16(pa, ld_contig16(vp + 48 * TSEQ), o3);
        asm volatile("s_wait_dscnt 0" ::: "memory");       // guard P WAR for next iter
    }

    // ---- final 1/l scale, store O as bf16 [B,T,C]
    u16* orow = ob + (size_t)(b * TSEQ + q0) * CDIM + h * HDIM;
#pragma unroll
    for (int r = 0; r < 8; ++r) {
        float inv = 1.0f / lrun[r];
        size_t base = (size_t)(r + hsel * 8) * CDIM + row;
        orow[base]      = cvt_bf16(o0[r] * inv);
        orow[base + 16] = cvt_bf16(o1[r] * inv);
        orow[base + 32] = cvt_bf16(o2[r] * inv);
        orow[base + 48] = cvt_bf16(o3[r] * inv);
    }
}

// ---------------------------------------------------------------------------
extern "C" void kernel_launch(void* const* d_in, const int* in_sizes, int n_in,
                              void* d_out, int out_size, void* d_ws, size_t ws_size,
                              hipStream_t stream) {
    const float* x     = (const float*)d_in[0];   // [2,2048,1024]
    const float* w_qkv = (const float*)d_in[1];   // [3072,1024]
    const float* w_out = (const float*)d_in[2];   // [1024,1024]
    float* out = (float*)d_out;                   // [2,2048,1024] fp32

    const int nx  = BATCH * TSEQ * CDIM;          // 4194304
    const int nwq = 3 * CDIM * CDIM;              // 3145728
    const int nwo = CDIM * CDIM;                  // 1048576
    const int M   = BATCH * TSEQ;                 // 4096

    char* ws = (char*)d_ws;                       // ~59 MB used
    u16* xbf   = (u16*)(ws);                      //  8.0 MB
    u16* wqbf  = (u16*)(ws + 8388608);            //  6.0 MB
    u16* wobf  = (u16*)(ws + 14680064);           //  2.0 MB
    u16* qkvbf = (u16*)(ws + 16777216);           // 24.0 MB  [B,T,3C]
    u16* vtbf  = (u16*)(ws + 41943040);           //  8.0 MB  [B,H,D,T]
    u16* obf   = (u16*)(ws + 50331648);           //  8.0 MB  [B,T,C]

    // 1) bf16 casts of activations + weights
    cvt_f32_bf16<<<nx  / 256, 256, 0, stream>>>(x,     xbf, nx);
    cvt_f32_bf16<<<nwq / 256, 256, 0, stream>>>(w_qkv, wqbf, nwq);
    cvt_f32_bf16<<<nwo / 256, 256, 0, stream>>>(w_out, wobf, nwo);

    // 2) qkv = x @ w_qkv^T   (4096 x 3072, K=1024) -> bf16
    gemm_bf16_nt<<<dim3(M / 16, 3072 / 512), 128, 0, stream>>>(
        xbf, wqbf, nullptr, qkvbf, M, 3 * CDIM, CDIM, 1);

    // 3) V transpose for PV B-operand contiguity
    build_vt<<<nx / 256, 256, 0, stream>>>(qkvbf, vtbf, nx);

    // 4) fused causal flash attention -> bf16 O
    attn_fwd<<<dim3(TSEQ / 16, HEADS, BATCH), 32, 0, stream>>>(qkvbf, vtbf, obf);

    // 5) out = O @ w_out^T   (4096 x 1024, K=1024) -> fp32
    gemm_bf16_nt<<<dim3(M / 16, 1024 / 512), 128, 0, stream>>>(
        obf, wobf, out, nullptr, M, CDIM, CDIM, 0);
}
